// ImportanceSampledAttention_2336462209735
// MI455X (gfx1250) — matrix operand order repaired and checked by reading, not compiled, not run on hardware
//
#include <hip/hip_runtime.h>
#include <cstdint>

// ---------------------------------------------------------------------------
// ImportanceSampledAttention for MI455X (gfx1250, wave32, WMMA)
// Pipeline:
//   k0 x4 : WT = bf16(W^T) for Wq,Wk,Wv,Wo   (one-time pack, LDS transpose)
//   k1 x3 : Q/K = bf16(x@W+b) head-major; V written transposed [bh][dk][s]
//   k2    : scores=QK^T*scale (WMMA) -> softmax -> Bernoulli importance
//           sampling (counter-hash RNG) -> attn row (bf16)
//   k3    : ctx = attn @ V   [attn 64x32 strided tile via TDM -> LDS,
//                             V^T B-frags direct from global, WMMA]
//   k4    : out = ctx @ Wo + bo (f32)        [WMMA bf16, no LDS]
// Workspace (bytes): Qbf[0,1M) Kbf[1M,2M) VT[2M,3M) Attn[3M,11M) Ctx[11M,13M)
//                    WqT[13M,13.5M) WkT[13.5M,14M) WvT[14M,14.5M) WoT[14.5M,15M)
// ---------------------------------------------------------------------------

typedef __attribute__((ext_vector_type(16))) __bf16   v16bf;
typedef __attribute__((ext_vector_type(8)))  float    v8f;
typedef __attribute__((ext_vector_type(4)))  uint32_t v4u;
typedef __attribute__((ext_vector_type(8)))  int      v8i;
typedef __attribute__((ext_vector_type(4)))  int      v4i;

union BF16x16 {
    v16bf  v;
    __bf16 h[16];
    uint4  q[2];
};

static_assert(sizeof(__bf16) == 2, "bf16 must be 2 bytes");
static_assert(sizeof(v16bf) == 32, "v16bf must be 32 bytes");

__device__ __forceinline__ v8f wmma_bf16(v16bf a, v16bf b, v8f c) {
    return __builtin_amdgcn_wmma_f32_16x16x32_bf16(false, a, false, b,
                                                   (short)0, c, false, false);
}

// two 8-bf16 chunks [p,p+8) and [p+16,p+24) (16B-aligned) -> v16bf
__device__ __forceinline__ v16bf load16_contig(const __bf16* p) {
    BF16x16 r;
    const uint4* s = (const uint4*)p;
    r.q[0] = s[0];
    r.q[1] = s[2];
    return r.v;
}

// A-fragment (16x32 bf16): per-lane chunks [base,base+8) and [base+16,base+24)
__device__ __forceinline__ v16bf loadA_frag(const __bf16* row, int base) {
    BF16x16 r;
    r.q[0] = *(const uint4*)(row + base);
    r.q[1] = *(const uint4*)(row + base + 16);
    return r.v;
}

__device__ __forceinline__ v16bf loadA_frag_f32(const float* row, int base) {
    BF16x16 r;
#pragma unroll
    for (int i = 0; i < 8; ++i) r.h[i] = (__bf16)row[base + i];
#pragma unroll
    for (int i = 0; i < 8; ++i) r.h[8 + i] = (__bf16)row[base + 16 + i];
    return r.v;
}

// ---- wave32 reductions ----------------------------------------------------
__device__ __forceinline__ float wave_max(float v) {
#pragma unroll
    for (int o = 16; o > 0; o >>= 1) v = fmaxf(v, __shfl_xor(v, o, 32));
    return v;
}
__device__ __forceinline__ float wave_sum(float v) {
#pragma unroll
    for (int o = 16; o > 0; o >>= 1) v += __shfl_xor(v, o, 32);
    return v;
}
__device__ __forceinline__ float wave_prod(float v) {
#pragma unroll
    for (int o = 16; o > 0; o >>= 1) v *= __shfl_xor(v, o, 32);
    return v;
}
__device__ __forceinline__ int wave_sumi(int v) {
#pragma unroll
    for (int o = 16; o > 0; o >>= 1) v += __shfl_xor(v, o, 32);
    return v;
}

// counter-based hash RNG in [0,1)
__device__ __forceinline__ float rng01(uint32_t row, uint32_t n, uint32_t k) {
    uint32_t h = row * 0x9E3779B1u + n * 0x85EBCA77u + k * 0xC2B2AE3Du + 0x27220A95u;
    h ^= h >> 16; h *= 0x7FEB352Du;
    h ^= h >> 15; h *= 0x846CA68Bu;
    h ^= h >> 16;
    return (float)h * 2.3283064365386963e-10f;
}

// generic pointer -> 32-bit LDS byte offset (addrspace(3) ptrtoint)
__device__ __forceinline__ uint32_t lds_addr_u32(const void* p) {
    return (uint32_t)(uintptr_t)(__attribute__((address_space(3))) const void*)p;
}

// TDM 2D tile load: rows x dim0 elements (2-byte data), row stride stride0
// (elements). Builds D# group0/group1 per CDNA5 ISA 8.3/8.4. 6-arg builtin.
__device__ __forceinline__ void tdm_load_2d_bf16(uint32_t lds_byte_addr,
                                                 const void* gsrc,
                                                 uint32_t dim0, uint32_t rows,
                                                 uint32_t stride0) {
    uint64_t ga = (uint64_t)(uintptr_t)gsrc;
    v4u g0;
    g0.x = 1u;                                        // count=1, user mode
    g0.y = lds_byte_addr;                             // lds_addr [63:32]
    g0.z = (uint32_t)(ga & 0xFFFFFFFFu);              // global_addr lo
    g0.w = (uint32_t)((ga >> 32) & 0x01FFFFFFu) | (2u << 30);  // addr hi | type=2
    v8i g1;
    g1[0] = (int)(1u << 16);                          // data_size=1 (2B), mask=0
    g1[1] = (int)((dim0 & 0xFFFFu) << 16);            // tensor_dim0 lo16 @bits63:48
    g1[2] = (int)(((dim0 >> 16) & 0xFFFFu) | ((rows & 0xFFFFu) << 16)); // dim0 hi | dim1 lo
    g1[3] = (int)(((rows >> 16) & 0xFFFFu) | ((dim0 & 0xFFFFu) << 16)); // dim1 hi | tile_dim0
    g1[4] = (int)(rows & 0xFFFFu);                    // tile_dim1 | tile_dim2=0
    g1[5] = (int)stride0;                             // tensor_dim0_stride lo32
    g1[6] = 0;                                        // stride0 hi | stride1 lo
    g1[7] = 0;                                        // stride1 hi
    v4i z4 = {0, 0, 0, 0};
    v8i z8 = {0, 0, 0, 0, 0, 0, 0, 0};
    __builtin_amdgcn_tensor_load_to_lds(g0, g1, z4, z4, z8, 0);
}

// ---------------------------------------------------------------------------
// Kernel 0: WT[n][k] = bf16(W[k][n]), 512x512. grid(16,16) block(32,8).
// ---------------------------------------------------------------------------
__global__ __launch_bounds__(256) void pack_wT_kernel(const float* __restrict__ W,
                                                      unsigned short* __restrict__ Traw) {
    __shared__ float tile[32][33];
    __bf16* T = (__bf16*)Traw;
    const int k0 = blockIdx.x * 32, n0 = blockIdx.y * 32;
    const int tx = threadIdx.x, ty = threadIdx.y;
#pragma unroll
    for (int j = 0; j < 4; ++j)
        tile[ty + 8 * j][tx] = W[(size_t)(k0 + ty + 8 * j) * 512 + n0 + tx];
    __syncthreads();
#pragma unroll
    for (int j = 0; j < 4; ++j)
        T[(size_t)(n0 + ty + 8 * j) * 512 + k0 + tx] = (__bf16)tile[tx][ty + 8 * j];
}

// ---------------------------------------------------------------------------
// Kernel 1: Y = bf16(x @ W + b). B-frags straight from WT (no LDS).
// v_transpose==0: head-major [bh][s][dk]; ==1: transposed [bh][dk][s] (for V).
// grid(16,8) block(128): 64x64 tile per block, 16x64 strip per wave.
// ---------------------------------------------------------------------------
__global__ __launch_bounds__(128) void qkv_proj_kernel(const float* __restrict__ X,
                                                       const unsigned short* __restrict__ WTraw,
                                                       const float* __restrict__ bias,
                                                       unsigned short* __restrict__ outraw,
                                                       int v_transpose) {
    const __bf16* WT = (const __bf16*)WTraw;
    __bf16* out = (__bf16*)outraw;
    const int lane = threadIdx.x & 31, wave = threadIdx.x >> 5;
    const int half = lane >> 4, l15 = lane & 15, base = half * 8;
    const int m0 = blockIdx.x * 64 + wave * 16;
    const int n0 = blockIdx.y * 64;
    const float* xrow = X + (size_t)(m0 + l15) * 512;
    v8f acc[4] = {};
    for (int k0 = 0; k0 < 512; k0 += 32) {
        v16bf a = loadA_frag_f32(xrow + k0, base);
#pragma unroll
        for (int t = 0; t < 4; ++t) {
            const __bf16* wtrow = WT + (size_t)(n0 + t * 16 + l15) * 512 + k0 + half * 8;
            if (k0 + 32 < 512) __builtin_prefetch(wtrow + 32, 0, 1);
            acc[t] = wmma_bf16(a, load16_contig(wtrow), acc[t]);
        }
    }
#pragma unroll
    for (int t = 0; t < 4; ++t) {
#pragma unroll
        for (int v = 0; v < 8; ++v) {
            int mm = m0 + half * 8 + v;
            int n = n0 + t * 16 + l15;
            float val = acc[t][v] + bias[n];
            int bb = mm >> 9, s = mm & 511, hh = n >> 6, d = n & 63;
            if (v_transpose)
                out[((size_t)((bb * 8 + hh) * 64 + d)) * 512 + s] = (__bf16)val;
            else
                out[((size_t)((bb * 8 + hh) * 512 + s)) * 64 + d] = (__bf16)val;
        }
    }
}

// ---------------------------------------------------------------------------
// Kernel 2: scores (WMMA) -> softmax -> importance sampling -> attn (bf16).
// grid(S/32=16, H=8, B=2) block(64): 2 waves, 16 query rows per wave.
// ---------------------------------------------------------------------------
__global__ __launch_bounds__(64) void scores_sample_kernel(const unsigned short* __restrict__ Qraw,
                                                           const unsigned short* __restrict__ Kraw,
                                                           unsigned short* __restrict__ attnraw) {
    __shared__ float sc[32][512];  // 64 KB score/target strip
    const __bf16* Qbf = (const __bf16*)Qraw;
    const __bf16* Kbf = (const __bf16*)Kraw;
    __bf16* attnbf = (__bf16*)attnraw;
    const int lane = threadIdx.x & 31, wave = threadIdx.x >> 5;
    const int half = lane >> 4, l15 = lane & 15, base = half * 8;
    const int bh = blockIdx.z * 8 + blockIdx.y;
    const int qblk = blockIdx.x * 32;
    const int q0 = qblk + wave * 16;
    const __bf16* Qrow = Qbf + (size_t)(bh * 512 + q0 + l15) * 64;
    v16bf a0 = loadA_frag(Qrow, base);        // dk 0..31 chunk
    v16bf a1 = loadA_frag(Qrow + 32, base);   // dk 32..63 chunk

    for (int n0 = 0; n0 < 512; n0 += 16) {
        const __bf16* Krow = Kbf + (size_t)(bh * 512 + n0 + l15) * 64;
        v16bf b0 = load16_contig(Krow + half * 8);
        v16bf b1 = load16_contig(Krow + 32 + half * 8);
        v8f acc = {0.f, 0.f, 0.f, 0.f, 0.f, 0.f, 0.f, 0.f};
        acc = wmma_bf16(a0, b0, acc);
        acc = wmma_bf16(a1, b1, acc);
#pragma unroll
        for (int v = 0; v < 8; ++v)
            sc[wave * 16 + half * 8 + v][n0 + l15] = acc[v] * 0.125f;  // 1/sqrt(64)
    }
    __syncthreads();

    const float invSk = 1.0f / 512.0f;
    for (int rr = 0; rr < 16; ++rr) {
        const int r = wave * 16 + rr;
        float* row = sc[r];
        const uint32_t rowid = (uint32_t)(bh * 512 + qblk + r);
        // softmax (BETA = 1)
        float mx = -3.4e38f;
        for (int k = lane; k < 512; k += 32) mx = fmaxf(mx, row[k]);
        mx = wave_max(mx);
        float se = 0.f;
        for (int k = lane; k < 512; k += 32) se += __expf(row[k] - mx);
        se = wave_sum(se);
        const float inv = 1.0f / se;
        for (int k = lane; k < 512; k += 32) row[k] = __expf(row[k] - mx) * inv;
        // pass 1: per-sample importance weights
        float wkeep = 0.f;
        for (int n = 0; n < 16; ++n) {
            float pt = 1.f, pp = 1.f;
            int cnt = 0;
            for (int k = lane; k < 512; k += 32) {
                float t = row[k];
                float p = 0.5f * t + 0.5f * invSk;
                bool s = rng01(rowid, n, k) < p;
                pt *= s ? t : 1.f;
                pp *= s ? p : (1.f - p);
                cnt += s ? 1 : 0;
            }
            pt = wave_prod(pt);
            pp = wave_prod(pp);
            cnt = wave_sumi(cnt);
            float wv = (cnt > 0 && pp > 0.f) ? pt / fmaxf(pp, 1e-38f) : 0.f;
            if (lane == n) wkeep = wv;
        }
        float wst = wave_sum(lane < 16 ? wkeep : 0.f);
        float wn_self = (wst > 0.f) ? wkeep / fmaxf(wst, 1e-38f) : (1.0f / 16.0f);
        float wn[16];
#pragma unroll
        for (int n = 0; n < 16; ++n) wn[n] = __shfl(wn_self, n, 32);
        // pass 2: accumulate weighted sample mask, normalize over keys
        float wa[16];
        float wsum = 0.f;
#pragma unroll
        for (int j = 0; j < 16; ++j) {
            int k = lane + 32 * j;
            float t = row[k];
            float p = 0.5f * t + 0.5f * invSk;
            float accw = 0.f;
            for (int n = 0; n < 16; ++n)
                accw += (rng01(rowid, n, k) < p) ? wn[n] : 0.f;
            wa[j] = accw;
            wsum += accw;
        }
        wsum = wave_sum(wsum);
        __bf16* arow = attnbf + (size_t)rowid * 512;
        const float invw = 1.0f / fmaxf(wsum, 1e-38f);
#pragma unroll
        for (int j = 0; j < 16; ++j) {
            int k = lane + 32 * j;
            float val = (wsum > 0.f) ? wa[j] * invw : invSk;
            arow[k] = (__bf16)val;
        }
    }
}

// ---------------------------------------------------------------------------
// Kernel 3: ctx = attn @ V per (b,h). TDM pulls the strided 64x32 attn tile
// (row stride 512) into LDS; A-frags are contiguous ds_load_b128; B-frags are
// contiguous global loads from V^T [bh][dk][s]. grid(8,8,2) block(128).
// ---------------------------------------------------------------------------
__global__ __launch_bounds__(128) void context_kernel(const unsigned short* __restrict__ attnraw,
                                                      const unsigned short* __restrict__ VTraw,
                                                      unsigned short* __restrict__ ctxraw) {
    __shared__ __attribute__((aligned(16))) __bf16 At[64 * 32];  // [row][k] 4KB
    const __bf16* attnbf = (const __bf16*)attnraw;
    const __bf16* VT = (const __bf16*)VTraw;
    __bf16* ctx = (__bf16*)ctxraw;
    const int lane = threadIdx.x & 31, wave = threadIdx.x >> 5;
    const int half = lane >> 4, l15 = lane & 15, base = half * 8;
    const int b = blockIdx.z, h = blockIdx.y;
    const int bh = b * 8 + h;
    const int qblk = blockIdx.x * 64;
    const int q0 = qblk + wave * 16;
    const uint32_t at_lds = lds_addr_u32(&At[0]);
    v8f acc[4] = {};
    for (int k0 = 0; k0 < 512; k0 += 32) {
        if (threadIdx.x < 32) {
            tdm_load_2d_bf16(at_lds,
                             attnbf + (size_t)(bh * 512 + qblk) * 512 + k0,
                             /*dim0=*/32, /*rows=*/64, /*stride0=*/512);
            __builtin_amdgcn_s_wait_tensorcnt(0);
        }
        __syncthreads();
        v16bf a = loadA_frag(At + (wave * 16 + l15) * 32, base);
#pragma unroll
        for (int t = 0; t < 4; ++t) {
            const __bf16* vrow = VT + (size_t)(bh * 64 + t * 16 + l15) * 512 + k0 + half * 8;
            if (k0 + 32 < 512) __builtin_prefetch(vrow + 32, 0, 1);
            acc[t] = wmma_bf16(a, load16_contig(vrow), acc[t]);
        }
        __syncthreads();
    }
#pragma unroll
    for (int t = 0; t < 4; ++t) {
#pragma unroll
        for (int v = 0; v < 8; ++v) {
            int s = q0 + half * 8 + v;
            int d = t * 16 + l15;
            ctx[(size_t)(b * 512 + s) * 512 + h * 64 + d] = (__bf16)acc[t][v];
        }
    }
}

// ---------------------------------------------------------------------------
// Kernel 4: out = ctx @ Wo + bo (f32). B-frags straight from WoT (no LDS).
// grid(16,8) block(128).
// ---------------------------------------------------------------------------
__global__ __launch_bounds__(128) void out_proj_kernel(const unsigned short* __restrict__ Xraw,
                                                       const unsigned short* __restrict__ WTraw,
                                                       const float* __restrict__ bias,
                                                       float* __restrict__ out) {
    const __bf16* X = (const __bf16*)Xraw;
    const __bf16* WT = (const __bf16*)WTraw;
    const int lane = threadIdx.x & 31, wave = threadIdx.x >> 5;
    const int half = lane >> 4, l15 = lane & 15, base = half * 8;
    const int m0 = blockIdx.x * 64 + wave * 16;
    const int n0 = blockIdx.y * 64;
    const __bf16* xrow = X + (size_t)(m0 + l15) * 512;
    v8f acc[4] = {};
    for (int k0 = 0; k0 < 512; k0 += 32) {
        v16bf a = loadA_frag(xrow + k0, base);
#pragma unroll
        for (int t = 0; t < 4; ++t) {
            const __bf16* wtrow = WT + (size_t)(n0 + t * 16 + l15) * 512 + k0 + half * 8;
            if (k0 + 32 < 512) __builtin_prefetch(wtrow + 32, 0, 1);
            acc[t] = wmma_bf16(a, load16_contig(wtrow), acc[t]);
        }
    }
#pragma unroll
    for (int t = 0; t < 4; ++t) {
#pragma unroll
        for (int v = 0; v < 8; ++v) {
            int mm = m0 + half * 8 + v;
            int n = n0 + t * 16 + l15;
            out[(size_t)mm * 512 + n] = acc[t][v] + bias[n];
        }
    }
}

// ---------------------------------------------------------------------------
extern "C" void kernel_launch(void* const* d_in, const int* in_sizes, int n_in,
                              void* d_out, int out_size, void* d_ws, size_t ws_size,
                              hipStream_t stream) {
    (void)in_sizes; (void)n_in; (void)out_size; (void)ws_size;
    const float* x  = (const float*)d_in[0];
    const float* Wq = (const float*)d_in[1];
    const float* bq = (const float*)d_in[2];
    const float* Wk = (const float*)d_in[3];
    const float* bk = (const float*)d_in[4];
    const float* Wv = (const float*)d_in[5];
    const float* bv = (const float*)d_in[6];
    const float* Wo = (const float*)d_in[7];
    const float* bo = (const float*)d_in[8];
    float* out = (float*)d_out;

    char* ws = (char*)d_ws;
    unsigned short* Qbf  = (unsigned short*)(ws);
    unsigned short* Kbf  = (unsigned short*)(ws + (1u << 20));
    unsigned short* VT   = (unsigned short*)(ws + (2u << 20));
    unsigned short* Attn = (unsigned short*)(ws + (3u << 20));
    unsigned short* Ctx  = (unsigned short*)(ws + (11u << 20));
    unsigned short* WqT  = (unsigned short*)(ws + (13u << 20));
    unsigned short* WkT  = (unsigned short*)(ws + (13u << 20) + (512u << 10));
    unsigned short* WvT  = (unsigned short*)(ws + (14u << 20));
    unsigned short* WoT  = (unsigned short*)(ws + (14u << 20) + (512u << 10));

    dim3 tg(16, 16), tb(32, 8);
    pack_wT_kernel<<<tg, tb, 0, stream>>>(Wq, WqT);
    pack_wT_kernel<<<tg, tb, 0, stream>>>(Wk, WkT);
    pack_wT_kernel<<<tg, tb, 0, stream>>>(Wv, WvT);
    pack_wT_kernel<<<tg, tb, 0, stream>>>(Wo, WoT);

    dim3 gg(16, 8), bb(128);
    qkv_proj_kernel<<<gg, bb, 0, stream>>>(x, WqT, bq, Qbf, 0);
    qkv_proj_kernel<<<gg, bb, 0, stream>>>(x, WkT, bk, Kbf, 0);
    qkv_proj_kernel<<<gg, bb, 0, stream>>>(x, WvT, bv, VT, 1);
    scores_sample_kernel<<<dim3(16, 8, 2), dim3(64), 0, stream>>>(Qbf, Kbf, Attn);
    context_kernel<<<dim3(8, 8, 2), dim3(128), 0, stream>>>(Attn, VT, Ctx);
    out_proj_kernel<<<gg, bb, 0, stream>>>(Ctx, WoT, bo, out);
}
